// AnchorAttention_75402445848764
// MI455X (gfx1250) — matrix operand order repaired and checked by reading, not compiled, not run on hardware
//
#include <hip/hip_runtime.h>
#include <hip/hip_fp16.h>

typedef __attribute__((ext_vector_type(16))) _Float16 v16h;
typedef __attribute__((ext_vector_type(8)))  _Float16 v8h;
typedef __attribute__((ext_vector_type(8)))  float    v8f;
typedef __attribute__((vector_size(16)))     int      i32x4;

#define DIMD   1024
#define NHEADS 16
#define HDIM   64
#define SEQ    4096
#define ANCH   512
#define QROWS  (SEQ - ANCH)   // 3584
#define BATCH  2
#define ATTN_SCALE 0.125f     // 1/sqrt(64)

#if defined(__gfx1250__) && __has_builtin(__builtin_amdgcn_global_load_async_to_lds_b128)
#define USE_ASYNC_LDS 1
#else
#define USE_ASYNC_LDS 0
#endif

__device__ __forceinline__ void wait_async0() {
#if defined(__gfx1250__)
#if __has_builtin(__builtin_amdgcn_s_wait_asynccnt)
  __builtin_amdgcn_s_wait_asynccnt(0);
#else
  asm volatile("s_wait_asynccnt 0x0" ::: "memory");
#endif
#endif
}

// ---------------- WMMA fragment helpers (16x16x32 f16, contiguous k order) ----------------
// A-matrix 16x32: lane L holds row m=L&15; khalf=L>>4 selects K sub-blocks:
//   a[i] = A[m][ khalf*16 + i ]  -> two contiguous 8-half (16B) runs.
__device__ __forceinline__ v16h frag_a(const _Float16* base, int row0, int stride) {
  const int lane = threadIdx.x & 31;
  const _Float16* p = base + (size_t)(row0 + (lane & 15)) * stride + ((lane >> 4) << 4);
  v8h x0 = *(const v8h*)(p);
  v8h x1 = *(const v8h*)(p + 8);
  v16h r;
#pragma unroll
  for (int i = 0; i < 8; ++i) { r[i] = x0[i]; r[8 + i] = x1[i]; }
  return r;
}

// B-matrix 32x16: lane L holds col n=col0+(L&15); b[i] = B[k = (L>>4)*16 + i][n].
// Stored col-major (n-major, k contiguous) -> one contiguous 16-half (32B) run.
__device__ __forceinline__ v16h frag_b(const _Float16* base, int col0, int stride) {
  const int lane = threadIdx.x & 31;
  return *(const v16h*)(base + (size_t)(col0 + (lane & 15)) * stride + ((lane >> 4) << 4));
}

__device__ __forceinline__ v8f wmma16(v16h a, v16h b, v8f c) {
  return __builtin_amdgcn_wmma_f32_16x16x32_f16(false, a, false, b, (short)0, c,
                                                false, false);
}

// ---------------- Projection GEMM (128x64 tile, 2x2 WMMA tiles/wave) -----------------
// out[b][h][outRowOff+s][d] = f16(x @ W + bias)
__global__ void __launch_bounds__(256)
proj_gemm_kernel(const float* __restrict__ X, const float* __restrict__ W,
                 const float* __restrict__ bias, _Float16* __restrict__ out,
                 int rowsPerBatch, int srcRowOff, int outRowsPerHead, int outRowOff) {
  __shared__ __align__(64) _Float16 As[128 * 32];  // [m][k]
  __shared__ __align__(64) _Float16 Bs[64 * 32];   // [n][k] (transposed)
  const int tid  = threadIdx.x;
  const int wave = tid >> 5;
  const int lane = tid & 31;
  const int rt0  = (wave & 3) << 1;   // row tile base (8 tiles of 16)
  const int ct0  = (wave >> 2) << 1;  // col tile base (4 tiles of 16)
  const int rblk = blockIdx.x * 128;
  const int nblk = blockIdx.y * 64;

  v8f acc[4] = {{}, {}, {}, {}};

  for (int kt = 0; kt < DIMD / 32; ++kt) {
#pragma unroll
    for (int rep = 0; rep < 4; ++rep) {          // A: 128x32 fp32 -> f16
      int idx = tid + rep * 256;
      int r = idx >> 3, c4 = (idx & 7) << 2;
      int g = rblk + r;
      int bb = (g >= rowsPerBatch) ? 1 : 0;
      int s  = g - bb * rowsPerBatch;
      const float4 f = *(const float4*)(X + ((size_t)(bb * SEQ + srcRowOff + s)) * DIMD + kt * 32 + c4);
      _Float16* d = As + r * 32 + c4;
      d[0] = (_Float16)f.x; d[1] = (_Float16)f.y;
      d[2] = (_Float16)f.z; d[3] = (_Float16)f.w;
    }
#pragma unroll
    for (int rep = 0; rep < 2; ++rep) {          // B: 32x64 fp32 -> Bs[n][k]
      int idx = tid + rep * 256;
      int k = idx >> 4, n4 = (idx & 15) << 2;
      const float4 f = *(const float4*)(W + ((size_t)(kt * 32 + k)) * DIMD + nblk + n4);
      Bs[(n4 + 0) * 32 + k] = (_Float16)f.x;
      Bs[(n4 + 1) * 32 + k] = (_Float16)f.y;
      Bs[(n4 + 2) * 32 + k] = (_Float16)f.z;
      Bs[(n4 + 3) * 32 + k] = (_Float16)f.w;
    }
    if (kt + 1 < DIMD / 32)                      // stream next weight tile into L2/WGP$
      __builtin_prefetch(W + ((size_t)((kt + 1) * 32 + (tid >> 3))) * DIMD + nblk + ((tid & 7) << 3), 0, 3);
    __syncthreads();
    v16h a0 = frag_a(As, (rt0 + 0) * 16, 32);
    v16h a1 = frag_a(As, (rt0 + 1) * 16, 32);
    v16h b0 = frag_b(Bs, (ct0 + 0) * 16, 32);
    v16h b1 = frag_b(Bs, (ct0 + 1) * 16, 32);
    acc[0] = wmma16(a0, b0, acc[0]);
    acc[1] = wmma16(a0, b1, acc[1]);
    acc[2] = wmma16(a1, b0, acc[2]);
    acc[3] = wmma16(a1, b1, acc[3]);
    __syncthreads();
  }

#pragma unroll
  for (int rt = 0; rt < 2; ++rt) {
#pragma unroll
    for (int ct = 0; ct < 2; ++ct) {
      const v8f& acc2 = acc[rt * 2 + ct];
      int col = nblk + (ct0 + ct) * 16 + (lane & 15);
      float bvv = bias[col];
      int h = col >> 6, d = col & 63;
#pragma unroll
      for (int v = 0; v < 8; ++v) {
        int g = rblk + (rt0 + rt) * 16 + ((lane >> 4) << 3) + v;
        int bb = (g >= rowsPerBatch) ? 1 : 0;
        int s  = g - bb * rowsPerBatch;
        size_t o = ((size_t)((bb * NHEADS + h) * outRowsPerHead) + outRowOff + s) * HDIM + d;
        out[o] = (_Float16)(acc2[v] + bvv);
      }
    }
  }
}

// ---------------- Flash attention: 1 block = (b,h) x 128 query rows ------------------
__global__ void __launch_bounds__(256)
attn_kernel(const _Float16* __restrict__ Qc, const _Float16* __restrict__ Kc,
            const _Float16* __restrict__ Vc, _Float16* __restrict__ Ctx) {
  extern __shared__ _Float16 smem[];
  _Float16* Kl = smem;                    // [512][64]  (a-major, k contiguous)
  _Float16* Vt = smem + ANCH * HDIM;      // [64][512]  (d-major, a contiguous)
  _Float16* Ps = smem + 2 * ANCH * HDIM;  // per-wave [16][32] bounce buffer

  const int tid   = threadIdx.x;
  const int wave  = tid >> 5;
  const int lane  = tid & 31;
  const int qtile = blockIdx.x & 31;
  const int bh    = blockIdx.x >> 5;      // b*16 + h
  const size_t kvbase = (size_t)bh * ANCH * HDIM;

#if USE_ASYNC_LDS
  for (int i = tid; i < ANCH * HDIM / 8; i += 256) {   // K: async DMA straight to LDS
    __builtin_amdgcn_global_load_async_to_lds_b128(
        (__attribute__((address_space(1))) i32x4*)(Kc + kvbase + i * 8),
        (__attribute__((address_space(3))) i32x4*)(Kl + i * 8), 0, 0);
  }
#else
  for (int i = tid; i < ANCH * HDIM / 8; i += 256)     // K straight copy
    *(v8h*)(Kl + i * 8) = *(const v8h*)(Kc + kvbase + i * 8);
#endif
  for (int i = tid; i < ANCH * HDIM / 8; i += 256) {   // V transpose
    int a = i >> 3, d8 = (i & 7) << 3;
    v8h vv = *(const v8h*)(Vc + kvbase + (size_t)a * HDIM + d8);
#pragma unroll
    for (int j = 0; j < 8; ++j) Vt[(size_t)(d8 + j) * ANCH + a] = vv[j];
  }
#if USE_ASYNC_LDS
  wait_async0();
#endif
  __syncthreads();

  const int qrow = qtile * 128 + wave * 16;
  const size_t qbase = ((size_t)bh * SEQ + qrow) * HDIM;
  v16h qf0 = frag_a(Qc + qbase, 0, HDIM);        // head dims 0..31
  v16h qf1 = frag_a(Qc + qbase + 32, 0, HDIM);   // head dims 32..63

  v8f accO[4] = {{}, {}, {}, {}};
  float mrow[8], lrow[8];
#pragma unroll
  for (int v = 0; v < 8; ++v) { mrow[v] = -1e30f; lrow[v] = 0.f; }
  _Float16* myPs = Ps + wave * 16 * 32;

  for (int ac = 0; ac < ANCH / 32; ++ac) {
    v8f s0 = {}, s1 = {};
    s0 = wmma16(qf0, frag_b(Kl,      ac * 32,      HDIM), s0);
    s0 = wmma16(qf1, frag_b(Kl + 32, ac * 32,      HDIM), s0);
    s1 = wmma16(qf0, frag_b(Kl,      ac * 32 + 16, HDIM), s1);
    s1 = wmma16(qf1, frag_b(Kl + 32, ac * 32 + 16, HDIM), s1);

#pragma unroll
    for (int v = 0; v < 8; ++v) {            // online softmax per row
      float a0 = s0[v] * ATTN_SCALE, a1 = s1[v] * ATTN_SCALE;
      float m = fmaxf(a0, a1);
      m = fmaxf(m, __shfl_xor(m, 1, 32));
      m = fmaxf(m, __shfl_xor(m, 2, 32));
      m = fmaxf(m, __shfl_xor(m, 4, 32));
      m = fmaxf(m, __shfl_xor(m, 8, 32));
      float mnew = fmaxf(mrow[v], m);
      float corr = __expf(mrow[v] - mnew);
      float p0 = __expf(a0 - mnew);
      float p1 = __expf(a1 - mnew);
      float sl = p0 + p1;
      sl += __shfl_xor(sl, 1, 32);
      sl += __shfl_xor(sl, 2, 32);
      sl += __shfl_xor(sl, 4, 32);
      sl += __shfl_xor(sl, 8, 32);
      lrow[v] = lrow[v] * corr + sl;
      mrow[v] = mnew;
#pragma unroll
      for (int dt = 0; dt < 4; ++dt) accO[dt][v] *= corr;
      int mi = ((lane >> 4) << 3) + v;       // C-layout row -> P bounce buffer
      myPs[mi * 32 + (lane & 15)]      = (_Float16)p0;
      myPs[mi * 32 + 16 + (lane & 15)] = (_Float16)p1;
    }
    v16h pf = frag_a(myPs, 0, 32);           // P as A-fragment (16x32)
#pragma unroll
    for (int dt = 0; dt < 4; ++dt)
      accO[dt] = wmma16(pf, frag_b(Vt + ac * 32, dt * 16, ANCH), accO[dt]);
  }

  const int b = bh >> 4, h = bh & 15;
#pragma unroll
  for (int dt = 0; dt < 4; ++dt) {
#pragma unroll
    for (int v = 0; v < 8; ++v) {
      int r = qrow + ((lane >> 4) << 3) + v;
      int c = dt * 16 + (lane & 15);
      float o = accO[dt][v] / lrow[v];
      Ctx[((size_t)(b * SEQ + r)) * DIMD + h * HDIM + c] = (_Float16)o;
    }
  }
}

// ---------------- Output GEMM: d_out = Ctx(f16) @ Wo + bo (fp32) ---------------------
__global__ void __launch_bounds__(256)
out_gemm_kernel(const _Float16* __restrict__ Atx, const float* __restrict__ W,
                const float* __restrict__ bias, float* __restrict__ out) {
  __shared__ __align__(64) _Float16 As[128 * 32];
  __shared__ __align__(64) _Float16 Bs[64 * 32];
  const int tid  = threadIdx.x;
  const int wave = tid >> 5;
  const int lane = tid & 31;
  const int rt0  = (wave & 3) << 1;
  const int ct0  = (wave >> 2) << 1;
  const int rblk = blockIdx.x * 128;
  const int nblk = blockIdx.y * 64;
  v8f acc[4] = {{}, {}, {}, {}};

  for (int kt = 0; kt < DIMD / 32; ++kt) {
#pragma unroll
    for (int rep = 0; rep < 2; ++rep) {          // A: 128x32 halves, b128 copies
      int idx = tid + rep * 256;
      int r = idx >> 2, c8 = (idx & 3) << 3;
      *(v8h*)(As + r * 32 + c8) =
          *(const v8h*)(Atx + ((size_t)(rblk + r)) * DIMD + kt * 32 + c8);
    }
#pragma unroll
    for (int rep = 0; rep < 2; ++rep) {          // B: 32x64 fp32 -> Bs[n][k]
      int idx = tid + rep * 256;
      int k = idx >> 4, n4 = (idx & 15) << 2;
      const float4 f = *(const float4*)(W + ((size_t)(kt * 32 + k)) * DIMD + nblk + n4);
      Bs[(n4 + 0) * 32 + k] = (_Float16)f.x;
      Bs[(n4 + 1) * 32 + k] = (_Float16)f.y;
      Bs[(n4 + 2) * 32 + k] = (_Float16)f.z;
      Bs[(n4 + 3) * 32 + k] = (_Float16)f.w;
    }
    if (kt + 1 < DIMD / 32)
      __builtin_prefetch(W + ((size_t)((kt + 1) * 32 + (tid >> 3))) * DIMD + nblk + ((tid & 7) << 3), 0, 3);
    __syncthreads();
    v16h a0 = frag_a(As, (rt0 + 0) * 16, 32);
    v16h a1 = frag_a(As, (rt0 + 1) * 16, 32);
    v16h b0 = frag_b(Bs, (ct0 + 0) * 16, 32);
    v16h b1 = frag_b(Bs, (ct0 + 1) * 16, 32);
    acc[0] = wmma16(a0, b0, acc[0]);
    acc[1] = wmma16(a0, b1, acc[1]);
    acc[2] = wmma16(a1, b0, acc[2]);
    acc[3] = wmma16(a1, b1, acc[3]);
    __syncthreads();
  }

#pragma unroll
  for (int rt = 0; rt < 2; ++rt) {
#pragma unroll
    for (int ct = 0; ct < 2; ++ct) {
      const v8f& acc2 = acc[rt * 2 + ct];
      int col = nblk + (ct0 + ct) * 16 + (lane & 15);
      float bvv = bias[col];
#pragma unroll
      for (int v = 0; v < 8; ++v) {
        int g = rblk + (rt0 + rt) * 16 + ((lane >> 4) << 3) + v;
        out[(size_t)g * DIMD + col] = acc2[v] + bvv;
      }
    }
  }
}

// ---------------- Launch -------------------------------------------------------------
extern "C" void kernel_launch(void* const* d_in, const int* in_sizes, int n_in,
                              void* d_out, int out_size, void* d_ws, size_t ws_size,
                              hipStream_t stream) {
  (void)in_sizes; (void)n_in; (void)out_size; (void)ws_size;
  const float* x   = (const float*)d_in[0];
  const float* Wq  = (const float*)d_in[1];
  const float* bq  = (const float*)d_in[2];
  const float* Wk  = (const float*)d_in[3];
  const float* bk  = (const float*)d_in[4];
  const float* Wv  = (const float*)d_in[5];
  const float* bv  = (const float*)d_in[6];
  const float* Wqt = (const float*)d_in[7];
  const float* bqt = (const float*)d_in[8];
  const float* Wo  = (const float*)d_in[9];
  const float* bo  = (const float*)d_in[10];

  char* ws = (char*)d_ws;
  _Float16* Qc  = (_Float16*)ws;                                       // [B][H][4096][64]
  _Float16* Kc  = Qc + (size_t)BATCH * NHEADS * SEQ * HDIM;            // [B][H][512][64]
  _Float16* Vc  = Kc + (size_t)BATCH * NHEADS * ANCH * HDIM;           // [B][H][512][64]
  _Float16* Ctx = Vc + (size_t)BATCH * NHEADS * ANCH * HDIM;           // [B][4096][1024]

  dim3 blk(256);
  // combined_q: anchors@Wq -> rows [0,512); queries@Wqt -> rows [512,4096)
  proj_gemm_kernel<<<dim3(BATCH * ANCH / 128, DIMD / 64), blk, 0, stream>>>(
      x, Wq, bq, Qc, ANCH, 0, SEQ, 0);
  proj_gemm_kernel<<<dim3(BATCH * QROWS / 128, DIMD / 64), blk, 0, stream>>>(
      x, Wqt, bqt, Qc, QROWS, ANCH, SEQ, ANCH);
  proj_gemm_kernel<<<dim3(BATCH * ANCH / 128, DIMD / 64), blk, 0, stream>>>(
      x, Wk, bk, Kc, ANCH, 0, ANCH, 0);
  proj_gemm_kernel<<<dim3(BATCH * ANCH / 128, DIMD / 64), blk, 0, stream>>>(
      x, Wv, bv, Vc, ANCH, 0, ANCH, 0);

  size_t attn_lds = (size_t)(2 * ANCH * HDIM + 8 * 16 * 32) * sizeof(_Float16); // 136 KB
  attn_kernel<<<dim3(BATCH * NHEADS * (SEQ / 128)), blk, attn_lds, stream>>>(Qc, Kc, Vc, Ctx);

  out_gemm_kernel<<<dim3(BATCH * SEQ / 128, DIMD / 64), blk, 0, stream>>>(
      Ctx, Wo, bo, (float*)d_out);
}
